// IrrepsConvolution_32366873542779
// MI455X (gfx1250) — matrix operand order repaired and checked by reading, not compiled, not run on hardware
//
#include <hip/hip_runtime.h>
#include <hip/hip_bf16.h>

typedef __attribute__((ext_vector_type(16))) _Float16 v16h;
typedef __attribute__((ext_vector_type(8)))  _Float16 v8h;
typedef __attribute__((ext_vector_type(8)))  float    v8f;

#define WAVES 2   // 64-thread blocks: 2 waves, keeps static LDS at 56 KB

__device__ __forceinline__ float sspf(float v) {
    // softplus(v) - log(2) == log(0.5 + 0.5*exp(v))
    //  = max(v,0) + ln2 * log2(0.5 + 0.5 * 2^(-|v|*log2e))
    // log argument is in [0.5, 1] -> safe to use raw v_exp_f32/v_log_f32
    const float L2E = 1.4426950408889634f;
    const float LN2 = 0.6931471805599453f;
    float e = __builtin_amdgcn_exp2f(-fabsf(v) * L2E);
    float l = __builtin_amdgcn_logf(__builtin_fmaf(e, 0.5f, 0.5f));
    return fmaxf(v, 0.0f) + LN2 * l;
}

__device__ __forceinline__ v8f wmma16(v16h a, v16h b, v8f c) {
    return __builtin_amdgcn_wmma_f32_16x16x32_f16(
        /*neg_a=*/false, a, /*neg_b=*/false, b,
        /*c_mod=*/(short)0, c, /*reuse_a=*/false, /*reuse_b=*/false);
}

__device__ __forceinline__ void atomAddF32(float* p, float v) {
    __hip_atomic_fetch_add(p, v, __ATOMIC_RELAXED, __HIP_MEMORY_SCOPE_AGENT);
}

__global__ void zero_f32(float* __restrict__ p, int n) {
    int i = blockIdx.x * blockDim.x + threadIdx.x;
    if (i < n) p[i] = 0.0f;
}

__global__ void __launch_bounds__(WAVES * 32)
fused_irreps_conv(const float* __restrict__ x,
                  const float* __restrict__ attr,
                  const float* __restrict__ emb,
                  const int*   __restrict__ eidx,
                  const float* __restrict__ W1,
                  const float* __restrict__ W2,
                  const float* __restrict__ W3,
                  const float* __restrict__ den,
                  float* __restrict__ out,
                  int E)
{
    // B fragments for all three layers: frag f, lane l, 16 halves (32B/lane)
    __shared__ __align__(16) _Float16 wfrag[32][32][16];     // 32 KB
    __shared__ __align__(16) _Float16 hbuf[WAVES][16 * 64];  // per-wave hidden
    __shared__ __align__(16) float    wbuf[WAVES][16 * 160]; // per-wave w output

    const int tid = threadIdx.x;
    const int l   = tid & 31;
    const int lh  = l & 15;   // lane within half-wave = N column / M row
    const int hs  = l >> 4;   // lane half selects K window
    const int waveId = tid >> 5;

    const float s1 = 0.35355339059327373f; // 1/sqrt(8)
    const float s2 = 0.125f;               // 1/sqrt(64)

    // ---- stage weights as WMMA B fragments (wave 0 only), scales baked in ----
    if (tid < 32) {
        #pragma unroll
        for (int t = 0; t < 4; ++t)
            #pragma unroll
            for (int h = 0; h < 16; ++h) {
                int k = hs * 16 + h;                 // K in 0..31 (only 0..7 valid)
                float v = (k < 8) ? W1[k * 64 + t * 16 + lh] * s1 : 0.0f;
                wfrag[t][l][h] = (_Float16)v;
            }
        for (int t = 0; t < 4; ++t)
            for (int k2 = 0; k2 < 2; ++k2)
                #pragma unroll
                for (int h = 0; h < 16; ++h) {
                    int k = k2 * 32 + hs * 16 + h;
                    wfrag[4 + t * 2 + k2][l][h] =
                        (_Float16)(W2[k * 64 + t * 16 + lh] * s2);
                }
        for (int t = 0; t < 10; ++t)
            for (int k2 = 0; k2 < 2; ++k2)
                #pragma unroll
                for (int h = 0; h < 16; ++h) {
                    int k = k2 * 32 + hs * 16 + h;
                    wfrag[12 + t * 2 + k2][l][h] =
                        (_Float16)(W3[k * 160 + t * 16 + lh] * s2);
                }
    }
    __syncthreads();

    const float invden = 1.0f / den[0];
    const float c0   = 0.17677669529663687f * invden;  // 1/sqrt(32)
    const float c110 = 0.10206207261596575f * invden;  // 1/sqrt(32)/sqrt(3)
    const float c111 = 0.125f * invden;                // 1/8

    _Float16* hb = &hbuf[waveId][0];
    float*    wb = &wbuf[waveId][0];

    const int nTiles = E >> 4;                   // E divisible by 16
    const int wavesTotal = gridDim.x * (blockDim.x >> 5);

    for (int tile = blockIdx.x * WAVES + waveId; tile < nTiles; tile += wavesTotal) {
        const int ebase = tile << 4;

        // ---- A1: 16 edges x 8 emb features, zero-padded to 16x32 f16 ----
        v16h a1 = {};
        if (hs == 0) {
            const float* er = emb + (size_t)(ebase + lh) * 8;
            #pragma unroll
            for (int j = 0; j < 8; ++j) a1[j] = (_Float16)er[j];
        }

        // ---- layer 1: 16x8 @ 8x64, softplus, stash f16 row-major in LDS ----
        #pragma unroll
        for (int t = 0; t < 4; ++t) {
            v16h b = *(const v16h*)&wfrag[t][l][0];
            v8f c = {};
            c = wmma16(a1, b, c);
            #pragma unroll
            for (int g = 0; g < 8; ++g)
                hb[(g + 8 * hs) * 64 + t * 16 + lh] = (_Float16)sspf(c[g]);
        }
        asm volatile("s_wait_dscnt 0" ::: "memory");

        // ---- reload hidden as A fragments (two 16x32 K-halves) ----
        v16h a2[2];
        #pragma unroll
        for (int k2 = 0; k2 < 2; ++k2) {
            v8h lo = *(const v8h*)&hb[lh * 64 + k2 * 32 + hs * 16];
            v8h hi = *(const v8h*)&hb[lh * 64 + k2 * 32 + hs * 16 + 8];
            a2[k2] = __builtin_shufflevector(lo, hi,
                      0,1,2,3,4,5,6,7,8,9,10,11,12,13,14,15);
        }
        asm volatile("s_wait_dscnt 0" ::: "memory");

        // ---- layer 2: 16x64 @ 64x64 ----
        #pragma unroll
        for (int t = 0; t < 4; ++t) {
            v16h b0 = *(const v16h*)&wfrag[4 + t * 2 + 0][l][0];
            v16h b1 = *(const v16h*)&wfrag[4 + t * 2 + 1][l][0];
            v8f c = {};
            c = wmma16(a2[0], b0, c);
            c = wmma16(a2[1], b1, c);
            #pragma unroll
            for (int g = 0; g < 8; ++g)
                hb[(g + 8 * hs) * 64 + t * 16 + lh] = (_Float16)sspf(c[g]);
        }
        asm volatile("s_wait_dscnt 0" ::: "memory");

        v16h a3[2];
        #pragma unroll
        for (int k2 = 0; k2 < 2; ++k2) {
            v8h lo = *(const v8h*)&hb[lh * 64 + k2 * 32 + hs * 16];
            v8h hi = *(const v8h*)&hb[lh * 64 + k2 * 32 + hs * 16 + 8];
            a3[k2] = __builtin_shufflevector(lo, hi,
                      0,1,2,3,4,5,6,7,8,9,10,11,12,13,14,15);
        }
        asm volatile("s_wait_dscnt 0" ::: "memory");

        // ---- layer 3: 16x64 @ 64x160, f32 result w -> LDS ----
        #pragma unroll
        for (int t = 0; t < 10; ++t) {
            v16h b0 = *(const v16h*)&wfrag[12 + t * 2 + 0][l][0];
            v16h b1 = *(const v16h*)&wfrag[12 + t * 2 + 1][l][0];
            v8f c = {};
            c = wmma16(a3[0], b0, c);
            c = wmma16(a3[1], b1, c);
            #pragma unroll
            for (int g = 0; g < 8; ++g)
                wb[(g + 8 * hs) * 160 + t * 16 + lh] = c[g];
        }
        asm volatile("s_wait_dscnt 0" ::: "memory");

        // ---- tensor product + scatter: 2 lanes per edge, 16 channels each ----
        const int el = l >> 1;
        const int uo = (l & 1) << 4;
        const int e  = ebase + el;
        const int dst = eidx[e];         // edge_idx[0] row
        const int src = eidx[E + e];     // edge_idx[1] row
        const float* xs = x + (size_t)src * 128;     // L2-resident gather
        const float f0  = attr[(size_t)e * 4 + 0];
        const float g1x = attr[(size_t)e * 4 + 1];
        const float g1y = attr[(size_t)e * 4 + 2];
        const float g1z = attr[(size_t)e * 4 + 3];
        float* op = out + (size_t)dst * 352;
        const float* wrow = wb + el * 160;

        #pragma unroll 4
        for (int uu = 0; uu < 16; ++uu) {
            const int u = uo + uu;
            float x0 = xs[u];
            float ax = xs[32 + u * 3 + 0];
            float ay = xs[32 + u * 3 + 1];
            float az = xs[32 + u * 3 + 2];
            float wA = wrow[u];
            float wB = wrow[32 + u];
            float wC = wrow[64 + u];
            float wD = wrow[96 + u];
            float wE = wrow[128 + u];
            float dot = ax * g1x + ay * g1y + az * g1z;
            float crx = ay * g1z - az * g1y;
            float cry = az * g1x - ax * g1z;
            float crz = ax * g1y - ay * g1x;
            atomAddF32(op + u,       c0   * wA * x0 * f0);
            atomAddF32(op + 32 + u,  c110 * wD * dot);
            float pb = c0 * wB * x0;
            atomAddF32(op + 64  + u * 3 + 0, pb * g1x);
            atomAddF32(op + 64  + u * 3 + 1, pb * g1y);
            atomAddF32(op + 64  + u * 3 + 2, pb * g1z);
            float pc = c0 * wC * f0;
            atomAddF32(op + 160 + u * 3 + 0, pc * ax);
            atomAddF32(op + 160 + u * 3 + 1, pc * ay);
            atomAddF32(op + 160 + u * 3 + 2, pc * az);
            float pe = c111 * wE;
            atomAddF32(op + 256 + u * 3 + 0, pe * crx);
            atomAddF32(op + 256 + u * 3 + 1, pe * cry);
            atomAddF32(op + 256 + u * 3 + 2, pe * crz);
        }
    }
}

extern "C" void kernel_launch(void* const* d_in, const int* in_sizes, int n_in,
                              void* d_out, int out_size, void* d_ws, size_t ws_size,
                              hipStream_t stream) {
    (void)n_in; (void)d_ws; (void)ws_size;
    const float* x    = (const float*)d_in[0];
    const float* attr = (const float*)d_in[1];
    const float* emb  = (const float*)d_in[2];
    const int*   eidx = (const int*)d_in[3];
    const float* W1   = (const float*)d_in[4];
    const float* W2   = (const float*)d_in[5];
    const float* W3   = (const float*)d_in[6];
    const float* den  = (const float*)d_in[7];
    float* out = (float*)d_out;
    const int E = in_sizes[1] / 4;   // edge_attr is (E,4)

    // Output is accumulated with atomics: must zero every call (graph replays).
    zero_f32<<<(out_size + 255) / 256, 256, 0, stream>>>(out, out_size);

    int nTiles = E >> 4;
    int blocks = (nTiles + WAVES - 1) / WAVES;
    if (blocks > 2048) blocks = 2048;   // ~4096 waves, ~8 edge-tiles each
    fused_irreps_conv<<<blocks, WAVES * 32, 0, stream>>>(
        x, attr, emb, eidx, W1, W2, W3, den, out, E);
}
